// GatedCrossModalAttention_50448685858853
// MI455X (gfx1250) — compile-verified, matrix operand order checked by reading, not run on hardware
//
#include <hip/hip_runtime.h>

// ---------------- problem constants ----------------
#define MOD   4
#define DIM   256
#define NH    8
#define HDIM  32
#define DGATE 64
#define DFF   1024
#define EPS   1e-5f
#define QSCALE 0.17677669529663687f   // 1/sqrt(32)

// ---------------- WMMA types ----------------
typedef __attribute__((ext_vector_type(16))) __bf16 v16bf;
typedef __attribute__((ext_vector_type(8)))  float  v8f;
typedef __attribute__((ext_vector_type(4)))  unsigned int v4u;
typedef __attribute__((ext_vector_type(4)))  int v4i;
typedef __attribute__((ext_vector_type(8)))  int v8i;

#if defined(__gfx1250__) && __has_builtin(__builtin_amdgcn_tensor_load_to_lds)
#define USE_TDM 1
#endif

__device__ __forceinline__ unsigned short f2bf(float f) {
  unsigned int u = __builtin_bit_cast(unsigned int, f);
  u += 0x7fffu + ((u >> 16) & 1u);               // round-to-nearest-even
  return (unsigned short)(u >> 16);
}

union Frag { v16bf v; uint4 q[2]; };

// A-fragment (16x32 bf16) from LDS, row-major source.
// ISA 7.12.2 16-bit A layout: lane<16 holds rows 0-15 K={j+0..7, j+16..23},
// lanes 16-31 same rows K={8..15, 24..31}:  k_local = j + 8*grp + (j>=8)*8.
__device__ __forceinline__ v16bf load_frag_lds(const unsigned short* base,
                                               int rowStride, int k0, int lane) {
  const int row = lane & 15, grp = lane >> 4;
  const unsigned short* p = base + row * rowStride + k0 + grp * 8;
  Frag u;
  u.q[0] = *(const uint4*)(p);        // 8 halves: k_local = 8*grp + 0..7
  u.q[1] = *(const uint4*)(p + 16);   // 8 halves: k_local = 8*grp + 16..23
  return u.v;
}

// B-fragment (32x16 bf16) from pre-swizzled global weights: lane reads 16
// contiguous halves (two b128 loads); swizzle done by pack_weights below.
__device__ __forceinline__ v16bf load_frag_w(const unsigned short* tileBase, int lane) {
  const uint4* p = (const uint4*)(tileBase + lane * 16);
  Frag u;
  u.q[0] = p[0];
  u.q[1] = p[1];
  return u.v;
}

__device__ __forceinline__ v8f wmma_bf16(v16bf a, v16bf b, v8f c) {
  return __builtin_amdgcn_wmma_f32_16x16x32_bf16(false, a, false, b,
                                                 (short)0, c, false, false);
}

__device__ __forceinline__ v8f vzero8() {
  v8f z = {0.f, 0.f, 0.f, 0.f, 0.f, 0.f, 0.f, 0.f};
  return z;
}

// ---------------- LDS layout (dynamic shared), byte offsets ----------------
#define XBS 1032   // halves per row of x tile (1024 + pad)
#define AOS 264    // floats per row (256 + pad)
#define OBS 264    // halves per row (256 + pad)
#define HGS 68
#define HFS 520    // halves per row of FF hidden half-slab (512 + pad)

#define SM_XB   0        // ushort[16*1032] x tile (bf16)            33024 B
#define SM_XSTG 33024    // float [16*1024] TDM staging (transient)  65536 B
#define SM_AO   33024    // float [16*264]  attn_out -> r -> y       16896 B
#define SM_SC   49920    // float [16*8*4]  scores -> softmax A       2048 B
#define SM_OHB  51968    // ushort[16*264]  Oh (bf16)                 8448 B
#define SM_ABF  60416    // ushort[16*264]  attn_out bf16 -> y bf16   8448 B
#define SM_HG   68864    // float [16*68]   gate hidden               4352 B
#define SM_GS   73216    // float [16]      gate scalar                 64 B
#define SM_P1   73280    // float [256]     LN partial sum            1024 B
#define SM_P2   74304    // float [256]     LN partial sumsq          1024 B
#define SM_MU   75328    // float [16]                                  64 B
#define SM_RS   75392    // float [16]                                  64 B
#define SM_HF   75456    // ushort[16*520]  FF hidden half (bf16)    16640 B
#define SM_BYTES 98560   // max(92096, SM_XSTG + 65536)

// ---------------- packed weight offsets in d_ws (halves) ----------------
#define OFF_WQ 0
#define OFF_WK 262144
#define OFF_WV 524288
#define OFF_WO 786432
#define OFF_F1 1048576
#define OFF_F2 2097152
#define OFF_G1 3145728

// Convert W[m][N][K] (fp32, out-major as in reference einsums) into bf16
// WMMA-B tiles: [m][kt][nt][lane][16 halves], matching load_frag_w + the
// k_local map used by load_frag_lds.
__global__ __launch_bounds__(256) void pack_weights(const float* __restrict__ src,
                                                    unsigned short* __restrict__ dst,
                                                    int K, int N, int mcnt) {
  const int NT = N >> 4;
  const long per_m = (long)K * N;          // == KT*NT*512
  const long total = per_m * mcnt;
  for (long idx = (long)blockIdx.x * blockDim.x + threadIdx.x; idx < total;
       idx += (long)gridDim.x * blockDim.x) {
    long r   = idx;
    int  m   = (int)(r / per_m);  r -= (long)m * per_m;
    int  kt  = (int)(r / ((long)NT * 512));
    int  r2  = (int)(r - (long)kt * NT * 512);
    int  nt  = r2 >> 9;
    int  r3  = r2 & 511;
    int  lane = r3 >> 4, j = r3 & 15;
    int  gcol = nt * 16 + (lane & 15);
    int  grp  = lane >> 4;
    int  kl   = j + 8 * grp + (j >= 8 ? 8 : 0);
    int  gk   = kt * 32 + kl;
    float v = src[(long)m * N * K + (long)gcol * K + gk];
    dst[idx] = f2bf(v);
  }
}

// ---------------- fused gated cross-modal attention block ----------------
__global__ __launch_bounds__(256, 2) void gcma_fused(
    const float* __restrict__ x, const unsigned short* __restrict__ wp,
    const float* __restrict__ bq, const float* __restrict__ bk,
    const float* __restrict__ bv, const float* __restrict__ bo,
    const float* __restrict__ g1b, const float* __restrict__ g2w,
    const float* __restrict__ g2b,
    const float* __restrict__ lng, const float* __restrict__ lnb,
    const float* __restrict__ f1b, const float* __restrict__ f2b,
    float* __restrict__ outy, float* __restrict__ outg) {
  extern __shared__ __align__(16) unsigned char smem[];
  unsigned short* xb  = (unsigned short*)(smem + SM_XB);
  float*          AO  = (float*)(smem + SM_AO);
  float*          Sc  = (float*)(smem + SM_SC);
  unsigned short* Ohb = (unsigned short*)(smem + SM_OHB);
  unsigned short* ABf = (unsigned short*)(smem + SM_ABF);
  float*          Hg  = (float*)(smem + SM_HG);
  float*          gsc = (float*)(smem + SM_GS);
  float*          ps1 = (float*)(smem + SM_P1);
  float*          ps2 = (float*)(smem + SM_P2);
  float*          muS = (float*)(smem + SM_MU);
  float*          rsS = (float*)(smem + SM_RS);
  unsigned short* HF  = (unsigned short*)(smem + SM_HF);

  const int tid  = threadIdx.x;
  const int lane = tid & 31;
  const int wv   = tid >> 5;          // 8 waves; wave wv owns head wv
  const int col  = lane & 15;
  const int lg   = lane >> 4;
  const long b0  = (long)blockIdx.x * 16;

  // ---- stage x tile: 16 rows x (M*D=1024) fp32 -> bf16 LDS ----
#if defined(USE_TDM)
  {
    float* xstage = (float*)(smem + SM_XSTG);
    if (wv == 0) {
      // 2D Tensor DMA descriptor (ISA 8.3/8.4): tile 1024x16 fp32,
      // tensor_dim0_stride = 1024 elements, load into LDS staging.
      const unsigned long long ga =
          (unsigned long long)(const void*)(x + b0 * (MOD * DIM));
      const unsigned int lds =
          (unsigned int)(unsigned long long)(void*)xstage;  // LDS byte addr
      v4u g0;
      g0[0] = 1u;                                          // count=1 (valid)
      g0[1] = lds;                                         // lds_addr
      g0[2] = (unsigned int)ga;                            // global_addr lo
      g0[3] = (unsigned int)((ga >> 32) & 0x1FFFFFFu) | (2u << 30); // hi|type=2
      v8i g1;
      g1[0] = (int)(2u << 16);                 // data_size = 4 bytes
      g1[1] = (int)(1024u << 16);              // tensor_dim0[15:0] = 1024
      g1[2] = (int)(16u << 16);                // td0 hi=0 | tensor_dim1 = 16
      g1[3] = (int)(1024u << 16);              // td1 hi=0 | tile_dim0 = 1024
      g1[4] = (int)16u;                        // tile_dim1 = 16, tile_dim2 = 0
      g1[5] = (int)1024;                       // tensor_dim0_stride = 1024
      g1[6] = 0;                               // stride hi | td1_stride lo
      g1[7] = 0;                               // td1_stride hi
      v4i gz = {0, 0, 0, 0};
#if __clang_major__ >= 23
      v8i gz8 = {0, 0, 0, 0, 0, 0, 0, 0};
      __builtin_amdgcn_tensor_load_to_lds(g0, g1, gz, gz, gz8, 0);
#else
      __builtin_amdgcn_tensor_load_to_lds(g0, g1, gz, gz, 0);
#endif
      __builtin_amdgcn_s_wait_tensorcnt(0);
    }
    __syncthreads();
    const int row = tid >> 4;
    const int c0  = (tid & 15) * 64;
    const float* sp = xstage + row * 1024 + c0;
    unsigned short* dp = xb + row * XBS + c0;
#pragma unroll
    for (int i = 0; i < 64; i += 4) {
      float4 f = *(const float4*)(sp + i);
      dp[i + 0] = f2bf(f.x); dp[i + 1] = f2bf(f.y);
      dp[i + 2] = f2bf(f.z); dp[i + 3] = f2bf(f.w);
    }
  }
#else
  {
    const int row = tid >> 4;
    const int c0  = (tid & 15) * 64;
    const float* xp = x + (b0 + row) * (MOD * DIM) + c0;
    unsigned short* dp = xb + row * XBS + c0;
#pragma unroll
    for (int i = 0; i < 64; i += 4) {
      float4 f = *(const float4*)(xp + i);
      dp[i + 0] = f2bf(f.x); dp[i + 1] = f2bf(f.y);
      dp[i + 2] = f2bf(f.z); dp[i + 3] = f2bf(f.w);
    }
  }
#endif
  __syncthreads();

  for (int m = 0; m < MOD; ++m) {
    // ======== Q = x[:,m] * Wq[m]^T + bq, scaled; kept in registers ========
    v8f qacc[2];
#pragma unroll
    for (int ti = 0; ti < 2; ++ti) {
      const int t = wv * 2 + ti;
      v8f acc = vzero8();
      const unsigned short* wb = wp + OFF_WQ + (long)m * 65536;
      for (int kt = 0; kt < 8; ++kt) {
        if (kt < 7) __builtin_prefetch(wb + ((kt + 1) * 16 + t) * 512, 0, 1);
        v16bf a = load_frag_lds(xb, XBS, m * DIM + kt * 32, lane);
        v16bf b = load_frag_w(wb + (kt * 16 + t) * 512, lane);
        acc = wmma_bf16(a, b, acc);
      }
      const float bias = bq[m * DIM + t * 16 + col];
#pragma unroll
      for (int r = 0; r < 8; ++r) acc[r] = (acc[r] + bias) * QSCALE;
      qacc[ti] = acc;
    }
    Sc[tid] = 0.f; Sc[tid + 256] = 0.f;   // zero score accumulators (512)
    __syncthreads();

    // ======== K_n = x[:,n]*Wk[m]^T + bk for all n, reusing each B tile ====
    {
      v8f kacc[2][4];
#pragma unroll
      for (int ti = 0; ti < 2; ++ti)
#pragma unroll
        for (int n = 0; n < MOD; ++n) kacc[ti][n] = vzero8();
#pragma unroll
      for (int ti = 0; ti < 2; ++ti) {
        const int t = wv * 2 + ti;
        const unsigned short* wb = wp + OFF_WK + (long)m * 65536;
        for (int kt = 0; kt < 8; ++kt) {
          v16bf b = load_frag_w(wb + (kt * 16 + t) * 512, lane);  // load once
#pragma unroll
          for (int n = 0; n < MOD; ++n) {                         // reuse x4
            v16bf a = load_frag_lds(xb, XBS, n * DIM + kt * 32, lane);
            kacc[ti][n] = wmma_bf16(a, b, kacc[ti][n]);
          }
        }
      }
      // scores: per-lane partial dot with Q, reduced via ds_add_f32
#pragma unroll
      for (int ti = 0; ti < 2; ++ti) {
        const int t = wv * 2 + ti;
        const float bias = bk[m * DIM + t * 16 + col];
#pragma unroll
        for (int n = 0; n < MOD; ++n) {
#pragma unroll
          for (int r = 0; r < 8; ++r) {
            const int row = r + 8 * lg;
            const float part = (kacc[ti][n][r] + bias) * qacc[ti][r];
            atomicAdd(&Sc[(row * NH + wv) * MOD + n], part);
          }
        }
      }
    }
    __syncthreads();

    // ======== softmax over n (N=4) per (row, head) ========
    if (tid < 16 * NH) {
      float* s = &Sc[tid * MOD];
      float mx = fmaxf(fmaxf(s[0], s[1]), fmaxf(s[2], s[3]));
      float e0 = __expf(s[0] - mx), e1 = __expf(s[1] - mx);
      float e2 = __expf(s[2] - mx), e3 = __expf(s[3] - mx);
      float inv = 1.f / (e0 + e1 + e2 + e3);
      s[0] = e0 * inv; s[1] = e1 * inv; s[2] = e2 * inv; s[3] = e3 * inv;
    }
    __syncthreads();

    // ======== V_n = x[:,n]*Wv[m]^T + bv (B tile reused over n) -> Oh =======
    {
      v8f vacc[2][4];
#pragma unroll
      for (int ti = 0; ti < 2; ++ti)
#pragma unroll
        for (int n = 0; n < MOD; ++n) vacc[ti][n] = vzero8();
#pragma unroll
      for (int ti = 0; ti < 2; ++ti) {
        const int t = wv * 2 + ti;
        const unsigned short* wb = wp + OFF_WV + (long)m * 65536;
        for (int kt = 0; kt < 8; ++kt) {
          v16bf b = load_frag_w(wb + (kt * 16 + t) * 512, lane);
#pragma unroll
          for (int n = 0; n < MOD; ++n) {
            v16bf a = load_frag_lds(xb, XBS, n * DIM + kt * 32, lane);
            vacc[ti][n] = wmma_bf16(a, b, vacc[ti][n]);
          }
        }
      }
#pragma unroll
      for (int ti = 0; ti < 2; ++ti) {
        const int t = wv * 2 + ti;
        const float bias = bv[m * DIM + t * 16 + col];
#pragma unroll
        for (int r = 0; r < 8; ++r) {
          const int row = r + 8 * lg;
          float o = 0.f;
#pragma unroll
          for (int n = 0; n < MOD; ++n)
            o += Sc[(row * NH + wv) * MOD + n] * (vacc[ti][n][r] + bias);
          Ohb[row * OBS + t * 16 + col] = f2bf(o);
        }
      }
    }
    __syncthreads();

    // ======== attn_out = Oh * Wo[m]^T + bo ========
#pragma unroll
    for (int ti = 0; ti < 2; ++ti) {
      const int t = wv * 2 + ti;
      v8f acc = vzero8();
      const unsigned short* wb = wp + OFF_WO + (long)m * 65536;
      for (int kt = 0; kt < 8; ++kt) {
        v16bf a = load_frag_lds(Ohb, OBS, kt * 32, lane);
        v16bf b = load_frag_w(wb + (kt * 16 + t) * 512, lane);
        acc = wmma_bf16(a, b, acc);
      }
      const int gc = t * 16 + col;
      const float bias = bo[m * DIM + gc];
#pragma unroll
      for (int r = 0; r < 8; ++r) {
        const int row = r + 8 * lg;
        const float av = acc[r] + bias;
        AO[row * AOS + gc] = av;
        ABf[row * OBS + gc] = f2bf(av);
      }
    }
    __syncthreads();

    // ======== gate layer1: tanh([x ; attn_out] * g1w^T + g1b) (WMMA) ========
    if (wv < 4) {
      const int t = wv;                       // 4 output tiles of 16 (DGATE=64)
      v8f acc = vzero8();
      const unsigned short* wb = wp + OFF_G1 + (long)m * 32768;
      for (int kt = 0; kt < 16; ++kt) {       // K = 2D = 512
        v16bf a = (kt < 8) ? load_frag_lds(xb, XBS, m * DIM + kt * 32, lane)
                           : load_frag_lds(ABf, OBS, (kt - 8) * 32, lane);
        v16bf b = load_frag_w(wb + (kt * 4 + t) * 512, lane);
        acc = wmma_bf16(a, b, acc);
      }
      const float bias = g1b[m * DGATE + t * 16 + col];
#pragma unroll
      for (int r = 0; r < 8; ++r)
        Hg[(r + 8 * lg) * HGS + t * 16 + col] = tanhf(acc[r] + bias);
    }
    __syncthreads();

    // ======== gate layer2 + sigmoid ========
    if (tid < 16) {
      float s = g2b[m];
      for (int j = 0; j < DGATE; ++j) s += Hg[tid * HGS + j] * g2w[m * DGATE + j];
      const float gt = 1.f / (1.f + __expf(-s));
      gsc[tid] = gt;
      outg[(b0 + tid) * MOD + m] = gt;
    }
    __syncthreads();

    // ======== r = x + gate*attn_out (in place into AO); LN stats ========
    {
      const int row = tid >> 4, cs = (tid & 15) * 16;
      const float* xp = x + (b0 + row) * (MOD * DIM) + m * DIM + cs;
      const float gt = gsc[row];
      float s1 = 0.f, s2 = 0.f;
#pragma unroll
      for (int i = 0; i < 16; ++i) {
        const float rv = xp[i] + gt * AO[row * AOS + cs + i];
        AO[row * AOS + cs + i] = rv;
        s1 += rv; s2 += rv * rv;
      }
      ps1[tid] = s1; ps2[tid] = s2;
    }
    __syncthreads();
    if (tid < 16) {
      float s1 = 0.f, s2 = 0.f;
      for (int i = 0; i < 16; ++i) { s1 += ps1[tid * 16 + i]; s2 += ps2[tid * 16 + i]; }
      const float mu = s1 * (1.f / DIM);
      const float var = s2 * (1.f / DIM) - mu * mu;
      muS[tid] = mu; rsS[tid] = rsqrtf(var + EPS);
    }
    __syncthreads();
    {
      const int row = tid >> 4, cs = (tid & 15) * 16;
      const float mu = muS[row], rs = rsS[row];
#pragma unroll
      for (int i = 0; i < 16; ++i) {
        const int c = cs + i;
        const float yv = (AO[row * AOS + c] - mu) * rs * lng[m * DIM + c] + lnb[m * DIM + c];
        AO[row * AOS + c] = yv;          // y (fp32) for final residual
        ABf[row * OBS + c] = f2bf(yv);   // y (bf16) as FF input
      }
    }
    __syncthreads();

    // ======== FeedForward: y -> GELU(y*f1w^T+f1b) -> *f2w^T+f2b, 2 halves ====
    v8f f2acc[2];
    f2acc[0] = vzero8(); f2acc[1] = vzero8();
    for (int half = 0; half < 2; ++half) {
      // FF1: 32 hidden tiles this half (4 per wave), K = 256
#pragma unroll
      for (int i = 0; i < 4; ++i) {
        const int t  = wv * 4 + i;        // local tile in half
        const int tg = half * 32 + t;     // global tile in DFF/16
        v8f acc = vzero8();
        const unsigned short* wb = wp + OFF_F1 + (long)m * 262144;
        for (int kt = 0; kt < 8; ++kt) {
          v16bf a = load_frag_lds(ABf, OBS, kt * 32, lane);
          v16bf b = load_frag_w(wb + (kt * 64 + tg) * 512, lane);
          acc = wmma_bf16(a, b, acc);
        }
        const float bias = f1b[m * DFF + tg * 16 + col];
#pragma unroll
        for (int r = 0; r < 8; ++r) {
          const float v = acc[r] + bias;
          const float ge = 0.5f * v * (1.f + erff(v * 0.70710678118654752f)); // exact GELU
          HF[(r + 8 * lg) * HFS + t * 16 + col] = f2bf(ge);
        }
      }
      __syncthreads();
      // FF2 partial accumulation over this half's K = 512
#pragma unroll
      for (int ti = 0; ti < 2; ++ti) {
        const int t = wv * 2 + ti;
        const unsigned short* wb = wp + OFF_F2 + (long)m * 262144;
        for (int kt2 = 0; kt2 < 16; ++kt2) {
          const int ktg = half * 16 + kt2;
          v16bf a = load_frag_lds(HF, HFS, kt2 * 32, lane);
          v16bf b = load_frag_w(wb + (ktg * 16 + t) * 512, lane);
          f2acc[ti] = wmma_bf16(a, b, f2acc[ti]);
        }
      }
      __syncthreads();   // HF reused next half
    }
    // updated = y + ff -> global
#pragma unroll
    for (int ti = 0; ti < 2; ++ti) {
      const int t = wv * 2 + ti;
      const int gc = t * 16 + col;
      const float bias = f2b[m * DIM + gc];
#pragma unroll
      for (int r = 0; r < 8; ++r) {
        const int row = r + 8 * lg;
        outy[((b0 + row) * MOD + m) * DIM + gc] = f2acc[ti][r] + bias + AO[row * AOS + gc];
      }
    }
    __syncthreads();   // buffers reused next modality
  }
}

// ---------------- launcher ----------------
extern "C" void kernel_launch(void* const* d_in, const int* in_sizes, int n_in,
                              void* d_out, int out_size, void* d_ws, size_t ws_size,
                              hipStream_t stream) {
  (void)n_in; (void)out_size; (void)ws_size;
  const float* x   = (const float*)d_in[0];
  const float* Wq  = (const float*)d_in[1];
  const float* bq  = (const float*)d_in[2];
  const float* Wk  = (const float*)d_in[3];
  const float* bk  = (const float*)d_in[4];
  const float* Wv  = (const float*)d_in[5];
  const float* bv  = (const float*)d_in[6];
  const float* Wo  = (const float*)d_in[7];
  const float* bo  = (const float*)d_in[8];
  const float* g1w = (const float*)d_in[9];
  const float* g1b = (const float*)d_in[10];
  const float* g2w = (const float*)d_in[11];
  const float* g2b = (const float*)d_in[12];
  const float* lng = (const float*)d_in[13];
  const float* lnb = (const float*)d_in[14];
  const float* f1w = (const float*)d_in[15];
  const float* f1b = (const float*)d_in[16];
  const float* f2w = (const float*)d_in[17];
  const float* f2b = (const float*)d_in[18];

  unsigned short* wpk = (unsigned short*)d_ws;
  const int nB = in_sizes[0] / (MOD * DIM);

  // weight pre-pack (bf16, WMMA-B swizzle) — every call, deterministic
  pack_weights<<<1024, 256, 0, stream>>>(Wq,  wpk + OFF_WQ, DIM,  DIM,  MOD);
  pack_weights<<<1024, 256, 0, stream>>>(Wk,  wpk + OFF_WK, DIM,  DIM,  MOD);
  pack_weights<<<1024, 256, 0, stream>>>(Wv,  wpk + OFF_WV, DIM,  DIM,  MOD);
  pack_weights<<<1024, 256, 0, stream>>>(Wo,  wpk + OFF_WO, DIM,  DIM,  MOD);
  pack_weights<<<2048, 256, 0, stream>>>(f1w, wpk + OFF_F1, DIM,  DFF,  MOD);
  pack_weights<<<2048, 256, 0, stream>>>(f2w, wpk + OFF_F2, DFF,  DIM,  MOD);
  pack_weights<<<256,  256, 0, stream>>>(g1w, wpk + OFF_G1, 2 * DIM, DGATE, MOD);

  float* outy = (float*)d_out;
  float* outg = outy + (size_t)nB * MOD * DIM;   // outputs: (updated, gate)
  gcma_fused<<<nB / 16, 256, SM_BYTES, stream>>>(
      x, wpk, bq, bk, bv, bo, g1b, g2w, g2b, lng, lnb, f1b, f2b, outy, outg);
}
